// DynamicA8W8MoeFFN_26637387170059
// MI455X (gfx1250) — compile-verified, matrix operand order checked by reading, not run on hardware
//
#include <hip/hip_runtime.h>
#include <hip/hip_bf16.h>
#include <stdint.h>

// Problem constants (match reference)
#define TT   4096          // tokens
#define HH   4096          // hidden
#define II   1408          // intermediate
#define EE   16            // experts
#define TPE  256           // tokens per expert (uniform)

typedef int v8i __attribute__((ext_vector_type(8)));

#define KC         128     // K bytes staged per LDS stage (2 WMMA K-steps)
#define LDS_STRIDE 144     // 128B payload + 16B pad; 16B-aligned rows, r*36 mod 64 distinct

// Assemble one 16x64 int8 WMMA operand fragment from LDS (row-major, K-contiguous).
// Per ISA 8-bit A/B layout: lanes 0-15 take K-bytes {0..7,16..23,32..39,48..55},
// lanes 16-31 take {8..15,24..31,40..47,56..63} of their row.
__device__ __forceinline__ v8i lds_frag(const char* base, int row0, int kofs, int lane) {
    const char* p = base + (size_t)(row0 + (lane & 15)) * LDS_STRIDE + kofs + ((lane >> 4) << 3);
    union { v8i v; unsigned long long q[4]; } u;
    u.q[0] = *(const unsigned long long*)(p +  0);
    u.q[1] = *(const unsigned long long*)(p + 16);
    u.q[2] = *(const unsigned long long*)(p + 32);
    u.q[3] = *(const unsigned long long*)(p + 48);
    return u.v;
}

__device__ __forceinline__ unsigned pack_q4(float4 v, float inv) {
    unsigned r = 0;
#pragma unroll
    for (int j = 0; j < 4; ++j) {
        float f = (j == 0) ? v.x : (j == 1) ? v.y : (j == 2) ? v.z : v.w;
        float q = rintf(f * inv);
        q = fminf(fmaxf(q, -128.f), 127.f);
        r |= ((unsigned)((int)q) & 0xffu) << (8 * j);
    }
    return r;
}

// ---------------------------------------------------------------------------
// Kernel 1: per-token dynamic quant of x (fp32 -> int8), also zero h-amax slot
// ---------------------------------------------------------------------------
__global__ __launch_bounds__(256) void quant_x_kernel(const float* __restrict__ x,
                                                      int8_t* __restrict__ xq,
                                                      float* __restrict__ s1,
                                                      unsigned* __restrict__ amax_h) {
    const int tok  = blockIdx.x;
    const int tid  = threadIdx.x;
    const int lane = tid & 31;
    const int wave = tid >> 5;
    const float4* row4 = (const float4*)(x + (size_t)tok * HH);   // 1024 float4

    float m = 0.f;
#pragma unroll
    for (int i = tid; i < HH / 4; i += 256) {
        float4 v = row4[i];
        m = fmaxf(m, fmaxf(fmaxf(fabsf(v.x), fabsf(v.y)), fmaxf(fabsf(v.z), fabsf(v.w))));
    }
    for (int off = 16; off >= 1; off >>= 1) m = fmaxf(m, __shfl_xor(m, off, 32));

    __shared__ float red[8];
    if (lane == 0) red[wave] = m;
    __syncthreads();
    float mm = red[0];
#pragma unroll
    for (int w = 1; w < 8; ++w) mm = fmaxf(mm, red[w]);

    const float scale = mm * (1.0f / 127.0f);
    const float inv   = (scale > 0.f) ? (1.0f / scale) : 0.f;
    unsigned* q4 = (unsigned*)(xq + (size_t)tok * HH);
#pragma unroll
    for (int i = tid; i < HH / 4; i += 256) q4[i] = pack_q4(row4[i], inv);
    if (tid == 0) { s1[tok] = scale; amax_h[tok] = 0u; }
}

// ---------------------------------------------------------------------------
// Kernel 2: GEMM1 (int8 WMMA) + scale + SiLU*up*smooth -> h (fp32) + row amax
// Block tile M=128 x 32 gate cols (+ matching 32 up cols).
// 256 threads = 8 waves in 4(M)x2(N); wave tile 32x16 per matrix.
// K staged in 128B slabs, double-buffered global->reg->LDS.
// ---------------------------------------------------------------------------
__global__ __launch_bounds__(256, 1) void gemm1_silu_kernel(const int8_t* __restrict__ xq,
                                                            const float* __restrict__ s1,
                                                            const int8_t* __restrict__ w13,
                                                            const float* __restrict__ w13_scale,
                                                            const float* __restrict__ smooth,
                                                            float* __restrict__ h,
                                                            unsigned* __restrict__ amax_h) {
    __shared__ __align__(16) char Abuf[2][128 * LDS_STRIDE];
    __shared__ __align__(16) char Gbuf[2][ 32 * LDS_STRIDE];
    __shared__ __align__(16) char Ubuf[2][ 32 * LDS_STRIDE];

    const int e     = blockIdx.z;
    const int mTile = blockIdx.y;            // 0..1
    const int nBase = blockIdx.x * 32;       // over I (44 tiles)
    const int tid   = threadIdx.x;
    const int wave  = tid >> 5, lane = tid & 31;
    const int waveM = wave >> 1, waveN = wave & 1;  // 4 x 2
    const int half  = lane >> 4, nl = lane & 15;

    const int8_t* gA = xq  + ((size_t)e * TPE + mTile * 128) * HH;
    const int8_t* gG = w13 + (size_t)e * (2 * II) * HH + (size_t)nBase * HH;
    const int8_t* gU = gG  + (size_t)II * HH;

    // Copy geometry (16B chunks): A 128 rows x 8 chunks -> 4/thread,
    // Bg/Bu 32 rows x 8 chunks -> 1/thread each.
    int aRow[4], aCol[4];
#pragma unroll
    for (int c = 0; c < 4; ++c) { int id = tid * 4 + c; aRow[c] = id >> 3; aCol[c] = (id & 7) << 4; }
    const int bRow = tid >> 3, bCol = (tid & 7) << 4;   // 0..31 rows

    v8i accG[2] = {};
    v8i accU[2] = {};
    uint4 ra[4], rg, ru;

    // Prologue: load stage 0
#pragma unroll
    for (int c = 0; c < 4; ++c)
        ra[c] = *(const uint4*)(gA + (size_t)aRow[c] * HH + aCol[c]);
    rg = *(const uint4*)(gG + (size_t)bRow * HH + bCol);
    ru = *(const uint4*)(gU + (size_t)bRow * HH + bCol);

    const int NK = HH / KC;   // 32
#pragma unroll 2
    for (int k = 0; k < NK; ++k) {
        const int buf = k & 1;   // constant-folds under the x2 unroll
        // Commit stage k to LDS
#pragma unroll
        for (int c = 0; c < 4; ++c)
            *(uint4*)&Abuf[buf][aRow[c] * LDS_STRIDE + aCol[c]] = ra[c];
        *(uint4*)&Gbuf[buf][bRow * LDS_STRIDE + bCol] = rg;
        *(uint4*)&Ubuf[buf][bRow * LDS_STRIDE + bCol] = ru;
        __syncthreads();

        // Prefetch stage k+1 into registers while WMMAs consume stage k
        if (k + 1 < NK) {
            const int k0 = (k + 1) * KC;
#pragma unroll
            for (int c = 0; c < 4; ++c)
                ra[c] = *(const uint4*)(gA + (size_t)aRow[c] * HH + k0 + aCol[c]);
            rg = *(const uint4*)(gG + (size_t)bRow * HH + k0 + bCol);
            ru = *(const uint4*)(gU + (size_t)bRow * HH + k0 + bCol);
        }

#pragma unroll
        for (int s = 0; s < 2; ++s) {
            const int ko = s * 64;
            // Phase 1: gather ALL fragments for this K-substep
            v8i af[2], bg, bu;
            bg = lds_frag(Gbuf[buf], waveN * 16, ko, lane);
            bu = lds_frag(Ubuf[buf], waveN * 16, ko, lane);
#pragma unroll
            for (int i = 0; i < 2; ++i)
                af[i] = lds_frag(Abuf[buf], waveM * 32 + i * 16, ko, lane);
            __builtin_amdgcn_sched_barrier(0);
            // Phase 2: uninterrupted WMMA run, no two consecutive on same acc
#pragma unroll
            for (int i = 0; i < 2; ++i)
                accG[i] = __builtin_amdgcn_wmma_i32_16x16x64_iu8(true, af[i], true, bg, accG[i], false, false);
#pragma unroll
            for (int i = 0; i < 2; ++i)
                accU[i] = __builtin_amdgcn_wmma_i32_16x16x64_iu8(true, af[i], true, bu, accU[i], false, false);
            __builtin_amdgcn_sched_barrier(0);
        }
        __syncthreads();
    }

    // Epilogue: y = acc * w13_scale * s1 ; h = silu(g)*u*smooth ; amax per token
    const int n   = nBase + waveN * 16 + nl;
    const float gsc = w13_scale[(size_t)e * 2 * II + n];
    const float usc = w13_scale[(size_t)e * 2 * II + II + n];
    const float ssm = smooth[(size_t)e * II + n];

#pragma unroll
    for (int i = 0; i < 2; ++i) {
#pragma unroll
        for (int r = 0; r < 8; ++r) {
            const int t   = mTile * 128 + waveM * 32 + i * 16 + r + 8 * half;
            const int tok = e * TPE + t;
            const float sv = s1[tok];
            const float g = (float)accG[i][r] * gsc * sv;
            const float u = (float)accU[i][r] * usc * sv;
            const float sig = __builtin_amdgcn_rcpf(1.0f + __expf(-g));  // v_rcp_f32, no div expansion
            const float hv = g * sig * u * ssm;
            h[(size_t)tok * II + n] = hv;
            // max |h| across the 16 lanes holding this token row
            float m = fabsf(hv);
            m = fmaxf(m, __shfl_xor(m, 1, 32));
            m = fmaxf(m, __shfl_xor(m, 2, 32));
            m = fmaxf(m, __shfl_xor(m, 4, 32));
            m = fmaxf(m, __shfl_xor(m, 8, 32));
            if (nl == 0) atomicMax(&amax_h[tok], __float_as_uint(m));
        }
    }
}

// ---------------------------------------------------------------------------
// Kernel 3: per-token dynamic quant of h (fp32 -> int8) using precomputed amax
// ---------------------------------------------------------------------------
__global__ __launch_bounds__(256) void quant_h_kernel(const float* __restrict__ h,
                                                      const unsigned* __restrict__ amax_h,
                                                      int8_t* __restrict__ hq,
                                                      float* __restrict__ s2) {
    const int tok = blockIdx.x;
    const int tid = threadIdx.x;
    const float a = __uint_as_float(amax_h[tok]);
    const float scale = a * (1.0f / 127.0f);
    const float inv   = (scale > 0.f) ? (1.0f / scale) : 0.f;
    const float4* row4 = (const float4*)(h + (size_t)tok * II);   // 352 float4
    unsigned* q4 = (unsigned*)(hq + (size_t)tok * II);
    for (int i = tid; i < II / 4; i += 256) q4[i] = pack_q4(row4[i], inv);
    if (tid == 0) s2[tok] = scale;
}

// ---------------------------------------------------------------------------
// Kernel 4: GEMM2 (int8 WMMA) + scale -> out (fp32)
// Block tile M=128 x N=64. 256 threads = 8 waves in 4x2; wave tile 32x32.
// K=1408 in 128B slabs, double-buffered global->reg->LDS.
// ---------------------------------------------------------------------------
__global__ __launch_bounds__(256, 1) void gemm2_kernel(const int8_t* __restrict__ hq,
                                                       const float* __restrict__ s2,
                                                       const int8_t* __restrict__ w2,
                                                       const float* __restrict__ w2_scale,
                                                       float* __restrict__ out) {
    __shared__ __align__(16) char Abuf[2][128 * LDS_STRIDE];
    __shared__ __align__(16) char Bbuf[2][ 64 * LDS_STRIDE];

    const int e     = blockIdx.z;
    const int mTile = blockIdx.y;            // 0..1
    const int nBase = blockIdx.x * 64;       // over H (64 tiles)
    const int tid   = threadIdx.x;
    const int wave  = tid >> 5, lane = tid & 31;
    const int waveM = wave >> 1, waveN = wave & 1;  // 4 x 2
    const int half  = lane >> 4, nl = lane & 15;

    const int8_t* gA = hq + ((size_t)e * TPE + mTile * 128) * II;
    const int8_t* gB = w2 + (size_t)e * HH * II + (size_t)nBase * II;

    // A: 128 rows x 8 chunks -> 4/thread; B: 64 rows x 8 chunks -> 2/thread
    int aRow[4], aCol[4], bRow[2], bCol[2];
#pragma unroll
    for (int c = 0; c < 4; ++c) { int id = tid * 4 + c; aRow[c] = id >> 3; aCol[c] = (id & 7) << 4; }
#pragma unroll
    for (int c = 0; c < 2; ++c) { int id = tid * 2 + c; bRow[c] = id >> 3; bCol[c] = (id & 7) << 4; }

    v8i acc[2][2] = {};
    uint4 ra[4], rb[2];

#pragma unroll
    for (int c = 0; c < 4; ++c)
        ra[c] = *(const uint4*)(gA + (size_t)aRow[c] * II + aCol[c]);
#pragma unroll
    for (int c = 0; c < 2; ++c)
        rb[c] = *(const uint4*)(gB + (size_t)bRow[c] * II + bCol[c]);

    const int NK = II / KC;   // 11
#pragma unroll 2
    for (int k = 0; k < NK; ++k) {
        const int buf = k & 1;   // constant-folds under the x2 unroll
#pragma unroll
        for (int c = 0; c < 4; ++c)
            *(uint4*)&Abuf[buf][aRow[c] * LDS_STRIDE + aCol[c]] = ra[c];
#pragma unroll
        for (int c = 0; c < 2; ++c)
            *(uint4*)&Bbuf[buf][bRow[c] * LDS_STRIDE + bCol[c]] = rb[c];
        __syncthreads();

        if (k + 1 < NK) {
            const int k0 = (k + 1) * KC;
#pragma unroll
            for (int c = 0; c < 4; ++c)
                ra[c] = *(const uint4*)(gA + (size_t)aRow[c] * II + k0 + aCol[c]);
#pragma unroll
            for (int c = 0; c < 2; ++c)
                rb[c] = *(const uint4*)(gB + (size_t)bRow[c] * II + k0 + bCol[c]);
        }

#pragma unroll
        for (int s = 0; s < 2; ++s) {
            const int ko = s * 64;
            // Phase 1: gather ALL fragments for this K-substep
            v8i af[2], bf[2];
#pragma unroll
            for (int j = 0; j < 2; ++j)
                bf[j] = lds_frag(Bbuf[buf], waveN * 32 + j * 16, ko, lane);
#pragma unroll
            for (int i = 0; i < 2; ++i)
                af[i] = lds_frag(Abuf[buf], waveM * 32 + i * 16, ko, lane);
            __builtin_amdgcn_sched_barrier(0);
            // Phase 2: uninterrupted WMMA run, no two consecutive on same acc
#pragma unroll
            for (int i = 0; i < 2; ++i)
#pragma unroll
                for (int j = 0; j < 2; ++j)
                    acc[i][j] = __builtin_amdgcn_wmma_i32_16x16x64_iu8(true, af[i], true, bf[j], acc[i][j], false, false);
            __builtin_amdgcn_sched_barrier(0);
        }
        __syncthreads();
    }

#pragma unroll
    for (int i = 0; i < 2; ++i) {
#pragma unroll
        for (int j = 0; j < 2; ++j) {
            const int n = nBase + waveN * 32 + j * 16 + nl;
            const float wsc = w2_scale[(size_t)e * HH + n];
#pragma unroll
            for (int r = 0; r < 8; ++r) {
                const int tok = e * TPE + mTile * 128 + waveM * 32 + i * 16 + r + 8 * half;
                out[(size_t)tok * HH + n] = (float)acc[i][j][r] * wsc * s2[tok];
            }
        }
    }
}

// ---------------------------------------------------------------------------
extern "C" void kernel_launch(void* const* d_in, const int* in_sizes, int n_in,
                              void* d_out, int out_size, void* d_ws, size_t ws_size,
                              hipStream_t stream) {
    (void)in_sizes; (void)n_in; (void)out_size; (void)ws_size;

    const float*  x          = (const float*)d_in[0];
    const int8_t* w13        = (const int8_t*)d_in[1];
    const int8_t* w2         = (const int8_t*)d_in[2];
    const float*  w13_scale  = (const float*)d_in[3];
    const float*  smooth     = (const float*)d_in[4];
    const float*  w2_scale   = (const float*)d_in[5];
    // d_in[6] expert_tokens: unused (uniform TPE, as in reference)
    float* out = (float*)d_out;

    // Workspace layout (256B aligned slices)
    char* ws = (char*)d_ws;
    size_t off = 0;
    auto take = [&](size_t bytes) -> char* {
        char* p = ws + off;
        off = (off + bytes + 255) & ~(size_t)255;
        return p;
    };
    int8_t*   xq     = (int8_t*)  take((size_t)TT * HH);        // 16 MB
    int8_t*   hq     = (int8_t*)  take((size_t)TT * II);        // 5.5 MB
    float*    h      = (float*)   take((size_t)TT * II * 4);    // 22 MB
    float*    s1     = (float*)   take((size_t)TT * 4);
    float*    s2     = (float*)   take((size_t)TT * 4);
    unsigned* amax_h = (unsigned*)take((size_t)TT * 4);

    // 1) quantize x, zero amax
    quant_x_kernel<<<TT, 256, 0, stream>>>(x, xq, s1, amax_h);

    // 2) GEMM1 + SiLU*up*smooth -> h, per-token amax
    dim3 g1(II / 32, TPE / 128, EE);   // (44, 2, 16)
    gemm1_silu_kernel<<<g1, 256, 0, stream>>>(xq, s1, w13, w13_scale, smooth, h, amax_h);

    // 3) quantize h
    quant_h_kernel<<<TT, 256, 0, stream>>>(h, amax_h, hq, s2);

    // 4) GEMM2 -> out
    dim3 g2(HH / 64, TPE / 128, EE);   // (64, 2, 16)
    gemm2_kernel<<<g2, 256, 0, stream>>>(hq, s2, w2, w2_scale, out);
}
